// ChanceCalculator_62706522522023
// MI455X (gfx1250) — compile-verified, matrix operand order checked by reading, not run on hardware
//
#include <hip/hip_runtime.h>
#include <hip/hip_bf16.h>
#include <math.h>

// ---------------------------------------------------------------------------
// GRU scan, batch=1, HIDDEN=2048, SEQ=16384 on MI455X (gfx1250).
//  - W_hh converted once to bf16 (24MB) -> stays resident in 192MB L2.
//  - Persistent kernel: 64 blocks x 192 threads (6 waves). Block b owns
//    h rows [32b, 32b+32) and weight rows {32b..+31} + {2048..} + {4096..}
//    (6 row-tiles of 16). Each wave computes one tile's FULL K=2048 dots
//    with v_wmma_f32_16x16x32_bf16 -> no inter-block reduction -> exactly
//    ONE grid barrier per timestep (ping-pong h buffer by step parity).
//  - h(t-1) (4KB bf16) pulled into LDS each step via the Tensor Data Mover
//    (tensor_load_to_lds + s_wait_tensorcnt).
//  - K-loop: manually double-buffered, 2 chunks per iteration, kept ROLLED
//    (#pragma unroll 1) so the compiler can neither fully unroll nor hoist
//    the weight loads (which caused scratch spills in the previous version).
// ---------------------------------------------------------------------------

#define HID   2048
#define SEQL  16384
#define NBLK  64
#define TPB   192                     // 6 waves of 32
#define ROWS_PER_BLK (HID / NBLK)     // 32
#define NTILE 6

typedef __attribute__((ext_vector_type(16))) __bf16 v16bf;
typedef __attribute__((ext_vector_type(8)))  __bf16 v8bf;
typedef __attribute__((ext_vector_type(8)))  float  v8f;
typedef __attribute__((ext_vector_type(4)))  unsigned int u32x4;
typedef __attribute__((ext_vector_type(8)))  int    i32x8;
typedef __attribute__((ext_vector_type(4)))  int    i32x4;

union BF8 { uint4 u; v8bf h; };

__device__ __forceinline__ float sigf(float x) { return 1.0f / (1.0f + __expf(-x)); }

// Sense-reversing grid barrier in workspace: bar[0]=count, bar[1]=generation.
__device__ __forceinline__ void grid_barrier(unsigned int* bar) {
  __syncthreads();
  if (threadIdx.x == 0) {
    unsigned int* cnt = bar;
    unsigned int* gen = bar + 1;
    unsigned int g = __hip_atomic_load(gen, __ATOMIC_RELAXED, __HIP_MEMORY_SCOPE_AGENT);
    unsigned int a = __hip_atomic_fetch_add(cnt, 1u, __ATOMIC_ACQ_REL, __HIP_MEMORY_SCOPE_AGENT);
    if (a == (unsigned int)(NBLK - 1)) {
      __hip_atomic_store(cnt, 0u, __ATOMIC_RELAXED, __HIP_MEMORY_SCOPE_AGENT);
      __hip_atomic_fetch_add(gen, 1u, __ATOMIC_RELEASE, __HIP_MEMORY_SCOPE_AGENT);
    } else {
      while (__hip_atomic_load(gen, __ATOMIC_ACQUIRE, __HIP_MEMORY_SCOPE_AGENT) == g)
        __builtin_amdgcn_s_sleep(1);
    }
    __threadfence();
  }
  __syncthreads();
}

// fp32 -> bf16 weight conversion (runs once per launch; W_hh then lives in L2)
__global__ void cvt_kernel(const float* __restrict__ w,
                           unsigned short* __restrict__ o, int n) {
  int i = blockIdx.x * blockDim.x + threadIdx.x;
  int stride = gridDim.x * blockDim.x;
  for (; i < n; i += stride) {
    __bf16 v = (__bf16)w[i];
    o[i] = __builtin_bit_cast(unsigned short, v);
  }
}

__global__ void init_kernel(unsigned short* hglob, unsigned int* bar, float* acc) {
  for (int k = threadIdx.x; k < 2 * HID; k += blockDim.x) hglob[k] = 0;
  if (threadIdx.x == 0) { bar[0] = 0u; bar[1] = 0u; acc[0] = 0.0f; }
}

__device__ __forceinline__ v16bf cat16(v8bf lo, v8bf hi) {
  return __builtin_shufflevector(lo, hi,
      0, 1, 2, 3, 4, 5, 6, 7, 8, 9, 10, 11, 12, 13, 14, 15);
}

__global__ __launch_bounds__(TPB, 1)
void gru_kernel(const float* __restrict__ samples,
                const float* __restrict__ w_ih,
                const float* __restrict__ b_ih,
                const float* __restrict__ b_hh,
                const float* __restrict__ fc_w,
                const float* __restrict__ fc_b,
                const unsigned short* __restrict__ wbf_us,
                unsigned short* __restrict__ hglob,   // ping-pong: 2 x HID bf16
                unsigned int* __restrict__ bar,
                float* __restrict__ acc,
                float* __restrict__ out) {
  __shared__ unsigned short hLds[HID];      // bf16 h(t-1), filled by TDM
  __shared__ float ghLds[NTILE * 16];       // 96 gate pre-activations

  const int tid  = threadIdx.x;
  const int blk  = blockIdx.x;
  const int wave = tid >> 5;
  const int lane = tid & 31;
  const int half = lane >> 4;               // 0: K-lo half, 1: K-hi half
  const int sel  = half * 8;                // per-lane K sub-offset (elements)

  // Wave 'wave' owns weight rows [tileRow0, tileRow0+16).
  const int gate0    = blk * ROWS_PER_BLK;
  const int tileRow0 = (wave >> 1) * HID + gate0 + (wave & 1) * 16;
  const __bf16* wbf  = reinterpret_cast<const __bf16*>(wbf_us);
  const __bf16* arow = wbf + (size_t)(tileRow0 + (lane & 15)) * HID;

  // b_hh for the 8 rows this lane will emit (only lanes with lane%16==0 emit).
  float bh[8];
#pragma unroll
  for (int j = 0; j < 8; ++j) bh[j] = 0.0f;
  if ((lane & 15) == 0) {
    const int rb = tileRow0 + 8 * half;
#pragma unroll
    for (int j = 0; j < 8; ++j) bh[j] = b_hh[rb + j];
  }

  // Gate-thread private state (wave 0, lanes 0..31 <-> h rows gate0..gate0+31).
  float hreg = 0.0f;
  float wr0 = 0, wr1 = 0, br = 0, wz0 = 0, wz1 = 0, bz = 0, wn0 = 0, wn1 = 0, bn = 0;
  if (tid < 32) {
    const int ig = gate0 + tid;
    wr0 = w_ih[2 * ig + 0];              wr1 = w_ih[2 * ig + 1];              br = b_ih[ig];
    wz0 = w_ih[2 * (ig + HID) + 0];      wz1 = w_ih[2 * (ig + HID) + 1];      bz = b_ih[ig + HID];
    wn0 = w_ih[2 * (ig + 2 * HID) + 0];  wn1 = w_ih[2 * (ig + 2 * HID) + 1];  bn = b_ih[ig + 2 * HID];
  }

  const unsigned int ldsOff = (unsigned int)(uintptr_t)&hLds[0]; // flat low32 = LDS offset

  for (int t = 0; t < SEQL; ++t) {
    // ---- bring h(t-1) bf16 into LDS (Tensor Data Mover, wave 0 only) ----
#if __has_builtin(__builtin_amdgcn_tensor_load_to_lds) && __has_builtin(__builtin_amdgcn_s_wait_tensorcnt)
    if (wave == 0) {
      unsigned long long ga =
          (unsigned long long)(uintptr_t)(hglob + (size_t)(t & 1) * HID);
      u32x4 g0;
      g0[0] = 1u;                                   // count=1, user descriptor
      g0[1] = ldsOff;                               // lds_addr
      g0[2] = (unsigned int)ga;                     // global_addr[31:0]
      g0[3] = ((unsigned int)(ga >> 32) & 0x01FFFFFFu) | 0x80000000u; // [56:32] | type=2
      i32x8 g1;
      g1[0] = 0x00010000;                           // data_size=1 (2 bytes/elem)
      g1[1] = (int)((unsigned)HID << 16);           // tensor_dim0 = 2048 (low16)
      g1[2] = 0x00010000;                           // tensor_dim1 = 1
      g1[3] = (int)((unsigned)HID << 16);           // tile_dim0 = 2048
      g1[4] = 1;                                    // tile_dim1 = 1
      g1[5] = HID;                                  // tensor_dim0_stride = 2048
      g1[6] = 0; g1[7] = 0;
      i32x4 zz4; zz4[0] = zz4[1] = zz4[2] = zz4[3] = 0;
      i32x8 zz8;
#pragma unroll
      for (int j = 0; j < 8; ++j) zz8[j] = 0;
      // amdgpu-toolchain (clang-23) 6-arg form: (g0, g1, g2, g3, g4, cpol)
      __builtin_amdgcn_tensor_load_to_lds(g0, g1, zz4, zz4, zz8, 0);
      __builtin_amdgcn_s_wait_tensorcnt(0);
    }
#else
    for (int i = tid; i < HID; i += TPB)
      hLds[i] = hglob[(size_t)(t & 1) * HID + i];
#endif
    __syncthreads();

    // ---- 16-row tile x full K=2048 via WMMA bf16 ----
    // Manual 2-stage pipeline with explicit double buffers; loop kept rolled.
    v8f c = {0.f, 0.f, 0.f, 0.f, 0.f, 0.f, 0.f, 0.f};

    BF8 a0A, a1A, a0B, a1B;
    v8bf b0A, b1A, b0B, b1B;

    // preload chunk 0 -> buffer A
    a0A.u = *reinterpret_cast<const uint4*>(arow + sel);
    a1A.u = *reinterpret_cast<const uint4*>(arow + 16 + sel);
    b0A   = *reinterpret_cast<const v8bf*>(&hLds[sel]);
    b1A   = *reinterpret_cast<const v8bf*>(&hLds[16 + sel]);

#pragma unroll 1
    for (int kb = 0; kb < HID - 64; kb += 64) {
      // issue loads for chunk kb+32 -> buffer B, then consume buffer A (kb)
      a0B.u = *reinterpret_cast<const uint4*>(arow + kb + 32 + sel);
      a1B.u = *reinterpret_cast<const uint4*>(arow + kb + 48 + sel);
      b0B   = *reinterpret_cast<const v8bf*>(&hLds[kb + 32 + sel]);
      b1B   = *reinterpret_cast<const v8bf*>(&hLds[kb + 48 + sel]);
      c = __builtin_amdgcn_wmma_f32_16x16x32_bf16(
          false, cat16(a0A.h, a1A.h), false, cat16(b0A, b1A),
          (short)0, c, false, false);
      // issue loads for chunk kb+64 -> buffer A, then consume buffer B (kb+32)
      a0A.u = *reinterpret_cast<const uint4*>(arow + kb + 64 + sel);
      a1A.u = *reinterpret_cast<const uint4*>(arow + kb + 80 + sel);
      b0A   = *reinterpret_cast<const v8bf*>(&hLds[kb + 64 + sel]);
      b1A   = *reinterpret_cast<const v8bf*>(&hLds[kb + 80 + sel]);
      c = __builtin_amdgcn_wmma_f32_16x16x32_bf16(
          false, cat16(a0B.h, a1B.h), false, cat16(b0B, b1B),
          (short)0, c, false, false);
    }
    // tail: buffer A holds chunk HID-64; load and do chunk HID-32
    a0B.u = *reinterpret_cast<const uint4*>(arow + (HID - 32) + sel);
    a1B.u = *reinterpret_cast<const uint4*>(arow + (HID - 16) + sel);
    b0B   = *reinterpret_cast<const v8bf*>(&hLds[(HID - 32) + sel]);
    b1B   = *reinterpret_cast<const v8bf*>(&hLds[(HID - 16) + sel]);
    c = __builtin_amdgcn_wmma_f32_16x16x32_bf16(
        false, cat16(a0A.h, a1A.h), false, cat16(b0A, b1A),
        (short)0, c, false, false);
    c = __builtin_amdgcn_wmma_f32_16x16x32_bf16(
        false, cat16(a0B.h, a1B.h), false, cat16(b0B, b1B),
        (short)0, c, false, false);

    // All 16 columns of D are identical (B columns replicated h chunk):
    // lane 0 holds tile rows 0..7 in c[0..7], lane 16 holds rows 8..15.
    if ((lane & 15) == 0) {
      const int o = wave * 16 + 8 * half;
#pragma unroll
      for (int j = 0; j < 8; ++j) ghLds[o + j] = c[j] + bh[j];
    }
    __syncthreads();

    // ---- gate math for this block's 32 h elements (wave 0) ----
    if (tid < 32) {
      const float s0 = samples[2 * t + 0];
      const float s1 = samples[2 * t + 1];
      const float hr = ghLds[tid];
      const float hz = ghLds[32 + tid];
      const float hn = ghLds[64 + tid];
      const float xr = fmaf(wr0, s0, fmaf(wr1, s1, br));
      const float xz = fmaf(wz0, s0, fmaf(wz1, s1, bz));
      const float xn = fmaf(wn0, s0, fmaf(wn1, s1, bn));
      const float r = sigf(xr + hr);
      const float z = sigf(xz + hz);
      const float n = tanhf(fmaf(r, hn, xn));
      hreg = (1.0f - z) * n + z * hreg;
      hglob[(size_t)((t + 1) & 1) * HID + gate0 + tid] =
          __builtin_bit_cast(unsigned short, (__bf16)hreg);
      __threadfence();   // publish h(t) device-wide before arriving
    }

    grid_barrier(bar);   // the single per-step device sync
  }

  // ---- final: out = sigmoid(h . fc_w + fc_b) ----
  if (tid < 32) {
    float p = hreg * fc_w[gate0 + tid];
#pragma unroll
    for (int o = 16; o > 0; o >>= 1) p += __shfl_down(p, o, 32);
    if (tid == 0) atomicAdd(acc, p);
  }
  grid_barrier(bar);
  if (blk == 0 && tid == 0) {
    float a = __hip_atomic_load(acc, __ATOMIC_RELAXED, __HIP_MEMORY_SCOPE_AGENT);
    out[0] = sigf(a + fc_b[0]);
  }
}

extern "C" void kernel_launch(void* const* d_in, const int* in_sizes, int n_in,
                              void* d_out, int out_size, void* d_ws, size_t ws_size,
                              hipStream_t stream) {
  (void)in_sizes; (void)n_in; (void)out_size; (void)ws_size;
  const float* samples = (const float*)d_in[0];
  const float* w_ih    = (const float*)d_in[1];
  const float* w_hh    = (const float*)d_in[2];
  const float* b_ih    = (const float*)d_in[3];
  const float* b_hh    = (const float*)d_in[4];
  const float* fc_w    = (const float*)d_in[5];
  const float* fc_b    = (const float*)d_in[6];

  // Workspace layout: [W_hh bf16: 3H*H*2 = 24MB][h ping-pong: 2*H bf16][barrier][acc]
  char* ws = (char*)d_ws;
  unsigned short* wbf = (unsigned short*)ws;
  size_t off = (size_t)3 * HID * HID * sizeof(unsigned short);
  unsigned short* hglob = (unsigned short*)(ws + off);
  off += (size_t)2 * HID * sizeof(unsigned short);
  off = (off + 15) & ~(size_t)15;
  unsigned int* bar = (unsigned int*)(ws + off);
  off += 16;
  float* acc = (float*)(ws + off);

  const int n = 3 * HID * HID;
  cvt_kernel<<<1024, 256, 0, stream>>>(w_hh, wbf, n);
  init_kernel<<<1, 256, 0, stream>>>(hglob, bar, acc);
  gru_kernel<<<NBLK, TPB, 0, stream>>>(samples, w_ih, b_ih, b_hh, fc_w, fc_b,
                                       wbf, hglob, bar, acc, (float*)d_out);
}
